// WeightOnlyQuantizedLinear_27608049778857
// MI455X (gfx1250) — matrix operand order repaired and checked by reading, not compiled, not run on hardware
//
#include <hip/hip_runtime.h>

// ---------------------------------------------------------------------------
// WeightOnlyQuantizedLinear: y[32,28672] = x[32,8192] @ W^T + bias
// bf16 WMMA + on-the-fly int4 dequant (0x4300 bias trick + v_pk_fma_bf16).
// k is permuted inside each 64-wide scale block (sum-invariant) so weight
// loads are one b128 per lane and nibble extraction is shift + and_or only.
// A prep kernel writes x as bf16 into d_ws with the matching permutation.
// The per-block 4KB A tile is staged in LDS with global_load_async_to_lds
// (double-buffered, ASYNCcnt) since all 8 waves of a block share it; waves
// read fragments with ds_load_b128.
// ---------------------------------------------------------------------------

#define M_DIM   32
#define K_DIM   8192
#define N_DIM   28672
#define KWORDS  (K_DIM / 8)        // 1024 int32 per weight row
#define NBLK    (K_DIM / 64)       // 128 scale blocks per row
#define ZWORDS  (NBLK / 8)         // 16 qzeros words per row

typedef __attribute__((ext_vector_type(16))) __bf16    v16bf;
typedef __attribute__((ext_vector_type(8)))  float     v8f;
typedef __attribute__((ext_vector_type(4)))  unsigned  v4u;

union FragB16 {
    unsigned u[8];
    v4u      q[2];
    v16bf    b;
};

__device__ __forceinline__ unsigned bf16bits(float f) {
    unsigned u = __builtin_bit_cast(unsigned, f);
    u += 0x7FFFu + ((u >> 16) & 1u);          // round-to-nearest-even
    return u >> 16;
}

__device__ __forceinline__ unsigned pk_fma_bf16(unsigned a, unsigned b, unsigned c) {
    unsigned d;
    asm volatile("v_pk_fma_bf16 %0, %1, %2, %3" : "=v"(d) : "v"(a), "v"(b), "v"(c));
    return d;
}

// async copy 16B/lane: LDS[lds_addr] = MEM[saddr + voff]   (ASYNCcnt)
__device__ __forceinline__ void async_copy_b128(unsigned lds_addr, unsigned voff,
                                                const void* saddr) {
    asm volatile("global_load_async_to_lds_b128 %0, %1, %2"
                 :: "v"(lds_addr), "v"(voff), "s"(saddr) : "memory");
}

__device__ __forceinline__ void wait_asynccnt0() {
    asm volatile("s_wait_asynccnt 0x0" ::: "memory");
}

// ---------------------------------------------------------------------------
// Kernel 0: x (f32) -> bf16 in d_ws, permuted to match fragment loads.
// ws[m][blk][c] = x[m][64*blk + phi(c)]
//   c = z*32 + t*16 + ha*8 + r ;  phi = 16t + 8z + 32ha + ((r&1)*4 + (r>>1))
// ---------------------------------------------------------------------------
__global__ __launch_bounds__(256) void prep_x_kernel(const float* __restrict__ x,
                                                     unsigned short* __restrict__ ws) {
    int idx = blockIdx.x * 256 + threadIdx.x;          // 0 .. 262143
    int m   = idx >> 13;
    int rem = idx & 8191;
    int blk = rem >> 6;
    int c   = rem & 63;
    int z  = (c >> 5) & 1;
    int t  = (c >> 4) & 1;
    int ha = (c >> 3) & 1;
    int r  = c & 7;
    int phys = blk * 64 + t * 16 + z * 8 + ha * 32 + ((r & 1) * 4 + (r >> 1));
    ws[idx] = (unsigned short)bf16bits(x[m * K_DIM + phys]);
}

// ---------------------------------------------------------------------------
// Kernel 1: GEMM.  One wave per 16-wide N tile (8 tiles / block).  Two 16x16
// f32 accumulators cover M=32 and share every dequantized B fragment.
// ---------------------------------------------------------------------------
__global__ __launch_bounds__(256) void w4_gemm_kernel(const int* __restrict__ qw,
                                                      const float* __restrict__ scales,
                                                      const int* __restrict__ qzeros,
                                                      const float* __restrict__ bias,
                                                      const unsigned short* __restrict__ ws,
                                                      float* __restrict__ out) {
    __shared__ __align__(16) unsigned short smem[2][32][64];   // 2 x 4KB A tiles

    const int tid   = threadIdx.x;
    const int lane  = tid & 31;
    const int wave  = tid >> 5;
    const int ntile = blockIdx.x * 8 + wave;           // 0 .. 1791
    const int ln    = lane & 15;                       // output column within tile
    const int hl    = lane >> 4;                       // k-half of the wave

    const int n = ntile * 16 + ln;
    const int*   qrow = qw     + (long)n * KWORDS;
    const float* srow = scales + (long)n * NBLK;
    const int*   zrow = qzeros + (long)n * ZWORDS;

    // async-copy addressing: thread tid moves 16B: row = tid>>3, c8 = tid&7
    const unsigned lds_base = (unsigned)(unsigned long long)&smem[0][0][0];
    const unsigned lds_off  = lds_base + (unsigned)tid * 16u;          // within buf0
    const unsigned gv_base  = ((unsigned)(tid >> 3) * K_DIM + (unsigned)(tid & 7) * 8) * 2u;

    v8f acc0 = {};
    v8f acc1 = {};

    // prologue: stage block 0 into buffer 0
    async_copy_b128(lds_off, gv_base, ws);

    for (int b = 0; b < NBLK; ++b) {
        wait_asynccnt0();        // own 16B of block b has landed in LDS
        __syncthreads();         // all waves' parts visible; prev compute done

        // stage block b+1 (wrapped; targets the buffer nobody reads this iter)
        {
            const unsigned nb = (unsigned)(b + 1) & (NBLK - 1);
            async_copy_b128(lds_off + (((unsigned)(b + 1) & 1u) << 12),
                            gv_base + nb * 128u, ws);
        }

        // per-(n, block) scale / zero-point -> packed bf16 constants
        float    s   = srow[b];
        int      zpw = zrow[b >> 3];
        int      zp  = ((zpw >> ((b & 7) * 4)) & 0xF) + 8;       // ZP_BASE = 8
        unsigned spk = bf16bits(s) * 0x10001u;
        unsigned cpk = bf16bits(-(float)(128 + zp) * s) * 0x10001u;

        // one b128 per lane: 32 consecutive nibbles (k = 64b + 32*hl .. +31)
        v4u w = *(const v4u*)(qrow + b * 8 + hl * 4);
        __builtin_prefetch(qrow + b * 8 + hl * 4 + 64, 0, 0);    // ~8 blocks ahead

        const unsigned short(*abuf)[64] = smem[b & 1];

#pragma unroll
        for (int t = 0; t < 2; ++t) {
            const unsigned w0 = w[2 * t];
            const unsigned w1 = w[2 * t + 1];

            FragB16 bf;
#pragma unroll
            for (int i = 0; i < 4; ++i) {
                unsigned v0 = ((w0 >> (4 * i)) & 0x000F000Fu) | 0x43004300u;
                unsigned v1 = ((w1 >> (4 * i)) & 0x000F000Fu) | 0x43004300u;
                bf.u[i]     = pk_fma_bf16(v0, spk, cpk);         // (q - zp) * s
                bf.u[4 + i] = pk_fma_bf16(v1, spk, cpk);
            }

            const int off = t * 16 + hl * 8;
            FragB16 a0, a1;
            a0.q[0] = *(const v4u*)&abuf[ln][off];
            a0.q[1] = *(const v4u*)&abuf[ln][off + 32];
            a1.q[0] = *(const v4u*)&abuf[ln + 16][off];
            a1.q[1] = *(const v4u*)&abuf[ln + 16][off + 32];

            acc0 = __builtin_amdgcn_wmma_f32_16x16x32_bf16(
                       false, a0.b, false, bf.b, (short)0, acc0, false, false);
            acc1 = __builtin_amdgcn_wmma_f32_16x16x32_bf16(
                       false, a1.b, false, bf.b, (short)0, acc1, false, false);
        }
    }

    // epilogue: + bias, store.  C/D layout: N = ln, M = 8*hl + reg (per tile).
    const float bb = bias[n];
#pragma unroll
    for (int r = 0; r < 8; ++r) {
        out[(long)(8 * hl + r) * N_DIM + n]      = acc0[r] + bb;
        out[(long)(16 + 8 * hl + r) * N_DIM + n] = acc1[r] + bb;
    }
}

// ---------------------------------------------------------------------------
extern "C" void kernel_launch(void* const* d_in, const int* in_sizes, int n_in,
                              void* d_out, int out_size, void* d_ws, size_t ws_size,
                              hipStream_t stream) {
    const float* x       = (const float*)d_in[0];
    const int*   qweight = (const int*)  d_in[1];
    const float* scales  = (const float*)d_in[2];
    const int*   qzeros  = (const int*)  d_in[3];
    const float* bias    = (const float*)d_in[4];
    float*       out     = (float*)d_out;

    unsigned short* xws = (unsigned short*)d_ws;     // needs 32*8192*2 = 512 KB

    // Kernel 0: x -> permuted bf16 workspace (262144 elements)
    prep_x_kernel<<<(M_DIM * K_DIM) / 256, 256, 0, stream>>>(x, xws);

    // Kernel 1: 1792 N tiles, 8 waves (tiles) per 256-thread block
    w4_gemm_kernel<<<(N_DIM / 16) / 8, 256, 0, stream>>>(
        qweight, scales, qzeros, bias, xws, out);
}
